// S5SSM_56083682951457
// MI455X (gfx1250) — compile-verified, matrix operand order checked by reading, not run on hardware
//
#include <hip/hip_runtime.h>
#include <math.h>

// ---------------------------------------------------------------------------
// S5 SSM fused kernel for gfx1250 (MI455X).
//  - bf16 WMMA for all five GEMM chains, complex scan state in registers.
//  - All weight B-fragments (B_bar, C_tilde, W_out) are loop-invariant and
//    live in VGPRs: each lane owns a fixed matrix column, so it builds its
//    own fragments straight from global params (no LDS weight tables).
//  - Input tiles double-buffered in LDS; async global->LDS DMA when the
//    toolchain exposes it; prefetch warms L2 one tile further ahead.
// ---------------------------------------------------------------------------

typedef __bf16 bf16_t;
typedef __attribute__((ext_vector_type(16))) __bf16 v16bf;
typedef __attribute__((ext_vector_type(8)))  __bf16 v8bf;
typedef __attribute__((ext_vector_type(8)))  float  v8f;

#define L_SEQ      100
#define M_TOTAL    2048       // BSZ*HS*WS = 8*16*16
#define U_DIM      128
#define P_DIM      128
#define TILE_M     16
#define WG_THREADS 256        // 8 wave32 waves; each wave owns one 16-col N-tile

// ---- LDS memory map (bytes); dynamic LDS, 28 KB ----
#define OFF_INF0    0         // input tile f32 [16][128], ping
#define OFF_INF1    8192      // input tile f32 [16][128], pong
#define OFF_XRE     16384     // x_re tile bf16 [16][128]
#define OFF_XIM     20480     // x_im tile bf16
#define OFF_YA      24576     // gelu(y) tile bf16
#define LDS_BYTES   28672

// ---------------- fast activation helpers (avoid v_div_scale chains) -------
__device__ inline float fast_rcp(float x) {
#if __has_builtin(__builtin_amdgcn_rcpf)
  return __builtin_amdgcn_rcpf(x);
#else
  return 1.0f / x;
#endif
}
__device__ inline float fast_sigmoid(float x) {
  return fast_rcp(1.0f + __expf(-x));
}

// ---------------- async global->LDS staging (guarded) ----------------------
__device__ inline void stage_tile(const float* __restrict__ gsrc,
                                  float* __restrict__ ldst, int tid) {
#if __has_builtin(__builtin_amdgcn_load_to_lds)
  // gfx1250 LDS-DMA: per-lane b32 transfers on the async path.
#pragma unroll
  for (int j = 0; j < 8; ++j) {
    const int idx = tid + j * WG_THREADS;     // 2048 contiguous f32
    __builtin_amdgcn_load_to_lds(
        (void*)(gsrc + idx),
        (__attribute__((address_space(3))) void*)(ldst + idx),
        /*size=*/4, /*offset=*/0, /*aux=*/0);
  }
#else
  for (int i = tid; i < TILE_M * U_DIM; i += WG_THREADS) ldst[i] = gsrc[i];
#endif
}
__device__ inline void stage_wait() {
#if __has_builtin(__builtin_amdgcn_s_wait_asynccnt)
  __builtin_amdgcn_s_wait_asynccnt(0);
#endif
}

// A-fragment K index for element e of a 16x32 bf16 fragment (per ISA layout):
// lane l holds row/col (l&15); elems 0..7 = k0..k0+7, elems 8..15 = k0+16..23,
// k0 = 32*kstep + 8*(l>>4).
__device__ inline int frag_k(int lane, int kstep, int e) {
  return kstep * 32 + ((e & 8) ? 16 : 0) + ((lane >> 4) << 3) + (e & 7);
}

// Load a 16x32 bf16 A-fragment from a row-major [16][128] bf16 LDS tile.
__device__ inline v16bf lds_load_fragA(const bf16_t* tile, int lane, int kstep) {
  const int row = lane & 15;
  const int k0  = kstep * 32 + ((lane >> 4) << 3);
  const v8bf* p = (const v8bf*)(tile + row * 128 + k0);
  v8bf lo = p[0];               // k0 .. k0+7
  v8bf hi = p[2];               // k0+16 .. k0+23
  return __builtin_shufflevector(lo, hi, 0,1,2,3,4,5,6,7,8,9,10,11,12,13,14,15);
}

// Same, but source tile is f32; convert to bf16 while building the fragment.
__device__ inline v16bf lds_load_fragA_f32(const float* tile, int lane, int kstep) {
  const int row = lane & 15;
  const int k0  = kstep * 32 + ((lane >> 4) << 3);
  const float* p = tile + row * 128 + k0;
  v8f lo = *(const v8f*)(p);
  v8f hi = *(const v8f*)(p + 16);
  v8bf blo = __builtin_convertvector(lo, v8bf);
  v8bf bhi = __builtin_convertvector(hi, v8bf);
  return __builtin_shufflevector(blo, bhi, 0,1,2,3,4,5,6,7,8,9,10,11,12,13,14,15);
}

__device__ inline v8f wmma_bf16(v16bf a, v16bf b, v8f c) {
  // (neg_a, A, neg_b, B, c_mod, C, reuse_a, reuse_b)
  return __builtin_amdgcn_wmma_f32_16x16x32_bf16(false, a, false, b, (short)0, c,
                                                 false, false);
}

__global__ __launch_bounds__(WG_THREADS, 1)
void s5_fused_kernel(const float* __restrict__ in_seq,  // (L, 2048, 128)
                     const float* __restrict__ x0,      // (2048, 128)
                     const float* __restrict__ Lam_re,  // (128)
                     const float* __restrict__ Lam_im,  // (128)
                     const float* __restrict__ B_param, // (128,128,2)
                     const float* __restrict__ C_param, // (128,128,2)
                     const float* __restrict__ D_vec,   // (128)
                     const float* __restrict__ log_stp, // (128)
                     const float* __restrict__ W_out,   // (128,128)
                     const float* __restrict__ b_out,   // (128)
                     float* __restrict__ out_xlast,     // (2048,128,2)
                     float* __restrict__ out_ys)        // (L,2048,128)
{
  extern __shared__ char smem[];
  float*  inF0 = (float* )(smem + OFF_INF0);
  float*  inF1 = (float* )(smem + OFF_INF1);
  bf16_t* xreL = (bf16_t*)(smem + OFF_XRE);
  bf16_t* ximL = (bf16_t*)(smem + OFF_XIM);
  bf16_t* yA   = (bf16_t*)(smem + OFF_YA);

  const int tid  = threadIdx.x;
  const int lane = tid & 31;
  const int wave = tid >> 5;          // 0..7 : this wave's 16-wide N-tile
  const int half = lane >> 4;         // C/D layout: row = vgpr + 8*half
  const int m0   = blockIdx.x * TILE_M;

  // ---- per-lane scalar constants (this lane's fixed matrix column) ----
  const int pl = (wave << 4) + (lane & 15);   // p for state / u for outputs
  const float st   = __expf(log_stp[pl]);
  const float lre  = fminf(Lam_re[pl], -1e-4f);
  const float lami = Lam_im[pl];
  const float eg   = __expf(lre * st);
  const float lbr  = eg * __cosf(lami * st);  // Lambda_bar (re)
  const float lbi  = eg * __sinf(lami * st);  // Lambda_bar (im)
  const float d_u  = D_vec[pl];
  const float bo   = b_out[pl];
  // (Lambda_bar - 1) / Lambda for this lane's p
  const float rden = fast_rcp(lre * lre + lami * lami);
  const float gr   = ((lbr - 1.0f) * lre + lbi * lami) * rden;
  const float gi   = (lbi * lre - (lbr - 1.0f) * lami) * rden;

  // ---- build ALL loop-invariant B-operand fragments in registers ----
  // Each lane's fragment column is pl, so B/C read row pl and W_out column pl of the
  // param matrices; only the K index varies across the 16 elements.
  v16bf rB_re[4], rB_im[4], rC_re[4], rC_in[4], rW[4];
#pragma unroll
  for (int ks = 0; ks < 4; ++ks) {
    v16bf brf, bif, crf, cif, wf;
#pragma unroll
    for (int e = 0; e < 16; ++e) {
      const int k = frag_k(lane, ks, e);
      // Bu GEMM:  B[k=u][n=p]  = B_bar[p][u] = ((Lb-1)/Lam)_p * B_tilde[p][u]
      const float b0 = B_param[((pl << 7) + k) * 2 + 0];
      const float b1 = B_param[((pl << 7) + k) * 2 + 1];
      brf[e] = (bf16_t)(gr * b0 - gi * b1);
      bif[e] = (bf16_t)(gr * b1 + gi * b0);
      // C GEMM:   B[k=p][n=u]  = C_tilde[u][p]   (minus baked into imag part)
      crf[e] = (bf16_t)( C_param[((pl << 7) + k) * 2 + 0]);
      cif[e] = (bf16_t)(-C_param[((pl << 7) + k) * 2 + 1]);
      // gate GEMM: B[k=u][n=j] = W_out[u][j]
      wf[e] = (bf16_t)W_out[(k << 7) + pl];
    }
    rB_re[ks] = brf; rB_im[ks] = bif;
    rC_re[ks] = crf; rC_in[ks] = cif;
    rW[ks]    = wf;
  }

  // ---- init complex state from x0 (real) in accumulator layout ----
  v8f x_re = {}, x_im = {};
#pragma unroll
  for (int v = 0; v < 8; ++v) {
    const int row = v + (half << 3);
    x_re[v] = x0[((m0 + row) << 7) + pl];
    x_im[v] = 0.0f;
  }

  // prime the ping buffer with step 0's tile (contiguous 8KB block)
  const float* gbase = in_seq + (size_t)m0 * U_DIM;
  stage_tile(gbase, inF0, tid);
  stage_wait();
  __syncthreads();

  // =============================== time loop ===============================
  for (int l = 0; l < L_SEQ; ++l) {
    float* cur = (l & 1) ? inF1 : inF0;
    float* nxt = (l & 1) ? inF0 : inF1;

    // stream next tile into the other buffer (overlaps this step's WMMAs),
    // and warm L2 one tile further ahead.
    if (l + 1 < L_SEQ)
      stage_tile(gbase + (size_t)(l + 1) * M_TOTAL * U_DIM, nxt, tid);
    if (l + 2 < L_SEQ)
      __builtin_prefetch(gbase + (size_t)(l + 2) * M_TOTAL * U_DIM + tid * 8, 0, 0);

    // ---- Bu = u_t @ B_bar^T  (complex RHS -> two bf16 WMMA chains) ----
    v8f bu_re = {}, bu_im = {};
#pragma unroll
    for (int ks = 0; ks < 4; ++ks) {
      v16bf a = lds_load_fragA_f32(cur, lane, ks);
      bu_re = wmma_bf16(a, rB_re[ks], bu_re);
      bu_im = wmma_bf16(a, rB_im[ks], bu_im);
    }

    // ---- complex recurrence x = Lambda_bar * x + Bu (state in registers) --
#pragma unroll
    for (int v = 0; v < 8; ++v) {
      const float xr = x_re[v], xi = x_im[v];
      x_re[v] = lbr * xr - lbi * xi + bu_re[v];
      x_im[v] = lbr * xi + lbi * xr + bu_im[v];
    }

    // publish x tile to LDS as bf16 A-operand data (row-major 16x128)
#pragma unroll
    for (int v = 0; v < 8; ++v) {
      const int row = v + (half << 3);
      xreL[(row << 7) + pl] = (bf16_t)x_re[v];
      ximL[(row << 7) + pl] = (bf16_t)x_im[v];
    }
    __syncthreads();

    // ---- y = 2*Re(C x) = 2*(x_re C_re - x_im C_im) ----
    v8f acc = {};
#pragma unroll
    for (int ks = 0; ks < 4; ++ks) {
      v16bf ar = lds_load_fragA(xreL, lane, ks);
      v16bf ai = lds_load_fragA(ximL, lane, ks);
      acc = wmma_bf16(ar, rC_re[ks], acc);
      acc = wmma_bf16(ai, rC_in[ks], acc);   // minus baked into rC_in
    }

    // ---- pointwise: + D*u, tanh-gelu via sigmoid identity; publish g ----
    float g[8];
#pragma unroll
    for (int v = 0; v < 8; ++v) {
      const int row = v + (half << 3);
      const float y = 2.0f * acc[v] + d_u * cur[(row << 7) + pl];
      // 0.5*y*(1+tanh(c*(y+0.044715 y^3))) == y*sigmoid(2c*(y+0.044715 y^3))
      g[v] = y * fast_sigmoid(1.5957691216057308f * (y + 0.044715f * y * y * y));
      yA[(row << 7) + pl] = (bf16_t)g[v];
    }
    __syncthreads();

    // ---- gate: z = g @ W_out ; out = g * sigmoid(z + b_out) ----
    v8f zacc = {};
#pragma unroll
    for (int ks = 0; ks < 4; ++ks) {
      v16bf a = lds_load_fragA(yA, lane, ks);
      zacc = wmma_bf16(a, rW[ks], zacc);
    }
    float* orow = out_ys + (size_t)l * M_TOTAL * U_DIM + (size_t)m0 * U_DIM + pl;
#pragma unroll
    for (int v = 0; v < 8; ++v) {
      const int row = v + (half << 3);
      orow[row * U_DIM] = g[v] * fast_sigmoid(zacc[v] + bo);
    }

    // next iteration's cur is complete; also protect xreL/yA reuse
    stage_wait();
    __syncthreads();
  }

  // ---- final state x_last (re/im interleaved, (2048,128,2)) ----
#pragma unroll
  for (int v = 0; v < 8; ++v) {
    const int row = v + (half << 3);
    const size_t base = (((size_t)(m0 + row) << 7) + pl) * 2;
    out_xlast[base + 0] = x_re[v];
    out_xlast[base + 1] = x_im[v];
  }
}

// ---------------------------------------------------------------------------
extern "C" void kernel_launch(void* const* d_in, const int* in_sizes, int n_in,
                              void* d_out, int out_size, void* d_ws, size_t ws_size,
                              hipStream_t stream) {
  const float* in_seq  = (const float*)d_in[0];
  const float* x0      = (const float*)d_in[1];
  const float* Lam_re  = (const float*)d_in[2];
  const float* Lam_im  = (const float*)d_in[3];
  const float* B_param = (const float*)d_in[4];
  const float* C_param = (const float*)d_in[5];
  const float* D_vec   = (const float*)d_in[6];
  const float* log_stp = (const float*)d_in[7];
  const float* W_out   = (const float*)d_in[8];
  const float* b_out   = (const float*)d_in[9];

  float* out_x  = (float*)d_out;                         // (2048,128,2)
  float* out_ys = out_x + (size_t)M_TOTAL * P_DIM * 2;   // (L,2048,128)

  dim3 grid(M_TOTAL / TILE_M);   // 128 workgroups
  dim3 block(WG_THREADS);        // 8 wave32 waves
  s5_fused_kernel<<<grid, block, LDS_BYTES, stream>>>(
      in_seq, x0, Lam_re, Lam_im, B_param, C_param, D_vec, log_stp,
      W_out, b_out, out_x, out_ys);
}
